// PointNetSetAbstraction_82471962018515
// MI455X (gfx1250) — matrix operand order repaired and checked by reading, not compiled, not run on hardware
//
#include <hip/hip_runtime.h>
#include <hip/hip_bf16.h>

#define NPOINT   1024
#define NSAMPLE  64
#define RADIUS2  0.04f
#define BN_EPS   1e-5f
#define BATCH    8
#define NPTS     8192

// d_out layout (floats): new_xyz (8,3,1024) | new_points (8,128,1024) | new_seed (8,1024)
#define OUT_NXYZ 0
#define OUT_NP   24576
#define OUT_SEED 1073152

// ws layout (bytes)
#define WS_W0    0          // f16 [64][8]   (Cin 6 padded to 8)
#define WS_W1    1024       // f16 [64][64]
#define WS_W2    9216       // f16 [128][64]
#define WS_BIAS  25600      // f32 [256]  (64 L1 | 64 L2 | 128 L3)
#define WS_CENT  26624      // f32 [8][1024][3]
#define WS_FEAT  131072     // f16 [8192][64][8]

typedef _Float16 half8 __attribute__((ext_vector_type(8)));
typedef _Float16 v16h  __attribute__((ext_vector_type(16)));
typedef float    v8f   __attribute__((ext_vector_type(8)));

__device__ __forceinline__ v16h pack16(half8 lo, half8 hi) {
  v16h r;
#pragma unroll
  for (int i = 0; i < 8; ++i) { r[i] = lo[i]; r[8 + i] = hi[i]; }
  return r;
}
// A fragment: lane<16 -> row, K [k..k+7] and [k+16..k+23]; lane>=16 offsets +8
__device__ __forceinline__ v16h lds_fragA(const _Float16* p, int base) {
  half8 lo = *(const half8*)(p + base);
  half8 hi = *(const half8*)(p + base + 16);
  return pack16(lo, hi);
}
// B fragment: lane<16 -> col, K [k..k+7] and [k+16..k+23]; lane>=16 offsets +8
__device__ __forceinline__ v16h lds_fragB(const _Float16* p, int base) {
  half8 lo = *(const half8*)(p + base);
  half8 hi = *(const half8*)(p + base + 16);
  return pack16(lo, hi);
}
__device__ __forceinline__ v8f wmma16(v16h a, v16h b, v8f c) {
  return __builtin_amdgcn_wmma_f32_16x16x32_f16(false, a, false, b, (short)0, c,
                                                false, false);
}

// ---------------- prep: fold BN into f16 weights + f32 bias ----------------
__global__ void prep_kernel(const float* w0, const float* b0, const float* g0,
                            const float* be0, const float* m0, const float* v0,
                            const float* w1, const float* b1, const float* g1,
                            const float* be1, const float* m1, const float* v1,
                            const float* w2, const float* b2, const float* g2,
                            const float* be2, const float* m2, const float* v2,
                            char* ws) {
  int tid = threadIdx.x;  // 256 threads = 64+64+128 out channels
  _Float16* W0f = (_Float16*)(ws + WS_W0);
  _Float16* W1f = (_Float16*)(ws + WS_W1);
  _Float16* W2f = (_Float16*)(ws + WS_W2);
  float* bias = (float*)(ws + WS_BIAS);
  if (tid < 64) {
    int o = tid;
    float sc = g0[o] * rsqrtf(v0[o] + BN_EPS);
    bias[tid] = (b0[o] - m0[o]) * sc + be0[o];
#pragma unroll
    for (int i = 0; i < 8; ++i)
      W0f[o * 8 + i] = (i < 6) ? (_Float16)(w0[o * 6 + i] * sc) : (_Float16)0.f;
  } else if (tid < 128) {
    int o = tid - 64;
    float sc = g1[o] * rsqrtf(v1[o] + BN_EPS);
    bias[tid] = (b1[o] - m1[o]) * sc + be1[o];
    for (int i = 0; i < 64; ++i) W1f[o * 64 + i] = (_Float16)(w1[o * 64 + i] * sc);
  } else {
    int o = tid - 128;
    float sc = g2[o] * rsqrtf(v2[o] + BN_EPS);
    bias[tid] = (b2[o] - m2[o]) * sc + be2[o];
    for (int i = 0; i < 64; ++i) W2f[o * 64 + i] = (_Float16)(w2[o * 64 + i] * sc);
  }
}

// ---------------- FPS: 1 block per batch, 1024 threads, 8 pts/thread -------
__global__ __launch_bounds__(1024) void fps_kernel(const float* __restrict__ xyz,
                                                   const int* __restrict__ seed,
                                                   float* __restrict__ out,
                                                   float* __restrict__ cent) {
  __shared__ float rv[32];
  __shared__ int ri[32];
  __shared__ int sWin;
  int b = blockIdx.x, tid = threadIdx.x;
  int lane = tid & 31, wid = tid >> 5;
  const float* xb = xyz + (size_t)b * 3 * NPTS;
  float px[8], py[8], pz[8], dist[8];
  int pn[8];
#pragma unroll
  for (int j = 0; j < 8; ++j) {
    int n = j * 1024 + tid;
    pn[j] = n;
    px[j] = xb[n];
    py[j] = xb[NPTS + n];
    pz[j] = xb[2 * NPTS + n];
    dist[j] = 1e10f;
  }
  int c = 0;
  for (int t = 0; t < NPOINT; ++t) {
    float cx = xb[c], cy = xb[NPTS + c], cz = xb[2 * NPTS + c];
    if (tid == 0) {
      out[OUT_NXYZ + (size_t)b * 3 * NPOINT + 0 * NPOINT + t] = cx;
      out[OUT_NXYZ + (size_t)b * 3 * NPOINT + 1 * NPOINT + t] = cy;
      out[OUT_NXYZ + (size_t)b * 3 * NPOINT + 2 * NPOINT + t] = cz;
      cent[((size_t)b * NPOINT + t) * 3 + 0] = cx;
      cent[((size_t)b * NPOINT + t) * 3 + 1] = cy;
      cent[((size_t)b * NPOINT + t) * 3 + 2] = cz;
      out[OUT_SEED + (size_t)b * NPOINT + t] = (float)seed[(size_t)b * NPTS + c];
    }
    float bv = -1.f;
    int bi = 0x7fffffff;
#pragma unroll
    for (int j = 0; j < 8; ++j) {
      float dx = px[j] - cx, dy = py[j] - cy, dz = pz[j] - cz;
      float d = dx * dx + dy * dy + dz * dz;
      dist[j] = fminf(dist[j], d);
      if (dist[j] > bv) { bv = dist[j]; bi = pn[j]; }
    }
#pragma unroll
    for (int off = 16; off; off >>= 1) {  // wave32 argmax, first-index ties
      float vo = __shfl_xor(bv, off, 32);
      int io = __shfl_xor(bi, off, 32);
      if (vo > bv || (vo == bv && io < bi)) { bv = vo; bi = io; }
    }
    if (lane == 0) { rv[wid] = bv; ri[wid] = bi; }
    __syncthreads();
    if (wid == 0) {
      float v2 = rv[lane];
      int i2 = ri[lane];
#pragma unroll
      for (int off = 16; off; off >>= 1) {
        float vo = __shfl_xor(v2, off, 32);
        int io = __shfl_xor(i2, off, 32);
        if (vo > v2 || (vo == v2 && io < i2)) { v2 = vo; i2 = io; }
      }
      if (lane == 0) sWin = i2;
    }
    __syncthreads();
    c = sWin;
  }
}

// -------- ball query + grouped feature build: 1 wave per query point -------
__global__ __launch_bounds__(256) void ball_kernel(const float* __restrict__ xyz,
                                                   const float* __restrict__ pts,
                                                   const float* __restrict__ cent,
                                                   _Float16* __restrict__ feat) {
  __shared__ int idxbuf[8][NSAMPLE];
  int lane = threadIdx.x & 31, wid = threadIdx.x >> 5;
  int g = blockIdx.x * 8 + wid;
  int b = g >> 10;
  const float* xb = xyz + (size_t)b * 3 * NPTS;
  const float* pb = pts + (size_t)b * 3 * NPTS;
  float cx = cent[(size_t)g * 3 + 0];
  float cy = cent[(size_t)g * 3 + 1];
  float cz = cent[(size_t)g * 3 + 2];
  int cnt = 0;  // wave-uniform
  for (int base = 0; base < NPTS && cnt < NSAMPLE; base += 32) {
    int n = base + lane;
    float dx = xb[n] - cx, dy = xb[NPTS + n] - cy, dz = xb[2 * NPTS + n] - cz;
    bool keep = (dx * dx + dy * dy + dz * dz) <= RADIUS2;
    unsigned mask = (unsigned)__ballot(keep);
    int pre = __popc(mask & ((1u << lane) - 1u));
    if (keep && (cnt + pre) < NSAMPLE) idxbuf[wid][cnt + pre] = n;
    cnt += __popc(mask);
  }
  __builtin_amdgcn_wave_barrier();
  int valid = cnt < NSAMPLE ? cnt : NSAMPLE;
  int first = idxbuf[wid][0];  // centroid itself always qualifies -> valid>=1
  for (int k = lane; k < NSAMPLE; k += 32)
    if (k >= valid) idxbuf[wid][k] = first;
  __builtin_amdgcn_wave_barrier();
  _Float16* fg = feat + (size_t)g * NSAMPLE * 8;
  for (int k = lane; k < NSAMPLE; k += 32) {
    int idx = idxbuf[wid][k];
    half8 h;
    h[0] = (_Float16)(xb[idx] - cx);
    h[1] = (_Float16)(xb[NPTS + idx] - cy);
    h[2] = (_Float16)(xb[2 * NPTS + idx] - cz);
    h[3] = (_Float16)pb[idx];
    h[4] = (_Float16)pb[NPTS + idx];
    h[5] = (_Float16)pb[2 * NPTS + idx];
    h[6] = (_Float16)0.f;
    h[7] = (_Float16)0.f;
    *(half8*)(fg + k * 8) = h;
  }
}

// -------- fused 3-layer MLP (WMMA f16) + max over K: 1 wave per group ------
#define ACT_STRIDE 72  // halves; 144B rows keep 16B alignment for b128 frags
#define SM_ACT 0       // 4 * 64 * 72 = 18432 halves
#define SM_W0 18432    // 512 halves
#define SM_W1 18944    // 4096 halves
#define SM_W2 23040    // 8192 halves
#define SM_BIASH 31232 // 256 f32 = 512 halves
#define SM_TOTAL 31744 // 63488 bytes

__global__ __launch_bounds__(128) void mlp_kernel(const _Float16* __restrict__ wts,
                                                  const float* __restrict__ biasg,
                                                  const _Float16* __restrict__ feat,
                                                  float* __restrict__ outp) {
  __shared__ __align__(16) _Float16 smem[SM_TOTAL];
  int tid = threadIdx.x, lane = tid & 31, wid = tid >> 5;
  {  // cooperative weight staging: 12800 halves = 1600 x 16B
    const uint4* src = (const uint4*)wts;
    uint4* dst = (uint4*)(smem + SM_W0);
    for (int i = tid; i < 1600; i += 128) dst[i] = src[i];
    float* bd = (float*)(smem + SM_BIASH);
    for (int i = tid; i < 256; i += 128) bd[i] = biasg[i];
  }
  __syncthreads();
  const float* biasf = (const float*)(smem + SM_BIASH);
  int g = blockIdx.x * 4 + wid;
  int b = g >> 10, s = g & 1023;
  int actb = SM_ACT + wid * 64 * ACT_STRIDE;
  int cl = lane & 15;
  bool lo16 = lane < 16;
  v8f zero8 = {};

  // ---- layer 1: feat (64xK=32, cols 6..31 zero) x W0^T -> act 64x64
  v16h a1[4];
#pragma unroll
  for (int mt = 0; mt < 4; ++mt) {
    v16h a = {};
    if (lo16) {  // lanes>=16 carry K=8..15 (all zero-padded)
      half8 f = *(const half8*)(feat + (size_t)g * 512 + (mt * 16 + lane) * 8);
#pragma unroll
      for (int i = 0; i < 8; ++i) a[i] = f[i];
    }
    a1[mt] = a;
  }
#pragma unroll
  for (int nt = 0; nt < 4; ++nt) {
    v16h bf = {};
    if (lo16) {
      half8 w = *(const half8*)(smem + SM_W0 + (nt * 16 + lane) * 8);
#pragma unroll
      for (int i = 0; i < 8; ++i) bf[i] = w[i];
    }
    float bb = biasf[nt * 16 + cl];
#pragma unroll
    for (int mt = 0; mt < 4; ++mt) {
      v8f acc = wmma16(a1[mt], bf, zero8);
      int row0 = mt * 16 + (lo16 ? 0 : 8);
#pragma unroll
      for (int r = 0; r < 8; ++r) {
        float v = fmaxf(acc[r] + bb, 0.f);
        smem[actb + (row0 + r) * ACT_STRIDE + nt * 16 + cl] = (_Float16)v;
      }
    }
  }
  __builtin_amdgcn_wave_barrier();

  // ---- layer 2: act 64x64 x W1^T -> act 64x64 (hold all 16 acc tiles)
  v8f acc2[16];
#pragma unroll
  for (int i = 0; i < 16; ++i) acc2[i] = zero8;
#pragma unroll
  for (int kk = 0; kk < 2; ++kk) {
    int k0 = kk * 32;
    v16h a[4];
#pragma unroll
    for (int mt = 0; mt < 4; ++mt)
      a[mt] = lds_fragA(smem, actb + (mt * 16 + cl) * ACT_STRIDE + k0 + (lo16 ? 0 : 8));
#pragma unroll
    for (int nt = 0; nt < 4; ++nt) {
      v16h bv = lds_fragB(smem, SM_W1 + (nt * 16 + cl) * 64 + k0 + (lo16 ? 0 : 8));
#pragma unroll
      for (int mt = 0; mt < 4; ++mt) acc2[nt * 4 + mt] = wmma16(a[mt], bv, acc2[nt * 4 + mt]);
    }
  }
  __builtin_amdgcn_wave_barrier();
#pragma unroll
  for (int nt = 0; nt < 4; ++nt) {
    float bb = biasf[64 + nt * 16 + cl];
#pragma unroll
    for (int mt = 0; mt < 4; ++mt) {
      int row0 = mt * 16 + (lo16 ? 0 : 8);
#pragma unroll
      for (int r = 0; r < 8; ++r) {
        float v = fmaxf(acc2[nt * 4 + mt][r] + bb, 0.f);
        smem[actb + (row0 + r) * ACT_STRIDE + nt * 16 + cl] = (_Float16)v;
      }
    }
  }
  __builtin_amdgcn_wave_barrier();

  // ---- layer 3: act 64x64 x W2^T (128 out) + ReLU + max over 64 rows
  v16h a3[2][4];
#pragma unroll
  for (int kk = 0; kk < 2; ++kk)
#pragma unroll
    for (int mt = 0; mt < 4; ++mt)
      a3[kk][mt] =
          lds_fragA(smem, actb + (mt * 16 + cl) * ACT_STRIDE + kk * 32 + (lo16 ? 0 : 8));
#pragma unroll
  for (int nt = 0; nt < 8; ++nt) {
    v8f acc[4];
#pragma unroll
    for (int i = 0; i < 4; ++i) acc[i] = zero8;
#pragma unroll
    for (int kk = 0; kk < 2; ++kk) {
      v16h bv = lds_fragB(smem, SM_W2 + (nt * 16 + cl) * 64 + kk * 32 + (lo16 ? 0 : 8));
#pragma unroll
      for (int mt = 0; mt < 4; ++mt) acc[mt] = wmma16(a3[kk][mt], bv, acc[mt]);
    }
    float bb = biasf[128 + nt * 16 + cl];
    float m = 0.f;  // ReLU floor == identity for max-of-relu
#pragma unroll
    for (int mt = 0; mt < 4; ++mt)
#pragma unroll
      for (int r = 0; r < 8; ++r) m = fmaxf(m, acc[mt][r] + bb);
    m = fmaxf(m, __shfl_xor(m, 16, 32));  // merge row halves (lanes c <-> c+16)
    if (lo16)
      outp[OUT_NP + (size_t)b * 128 * NPOINT + (size_t)(nt * 16 + lane) * NPOINT + s] = m;
  }
}

extern "C" void kernel_launch(void* const* d_in, const int* in_sizes, int n_in,
                              void* d_out, int out_size, void* d_ws, size_t ws_size,
                              hipStream_t stream) {
  const float* xyz = (const float*)d_in[0];
  const float* pts = (const float*)d_in[1];
  const int* seed = (const int*)d_in[2];
  const float* w0 = (const float*)d_in[3], *b0 = (const float*)d_in[4],
              *g0 = (const float*)d_in[5], *be0 = (const float*)d_in[6],
              *m0 = (const float*)d_in[7], *v0 = (const float*)d_in[8];
  const float* w1 = (const float*)d_in[9], *b1 = (const float*)d_in[10],
              *g1 = (const float*)d_in[11], *be1 = (const float*)d_in[12],
              *m1 = (const float*)d_in[13], *v1 = (const float*)d_in[14];
  const float* w2 = (const float*)d_in[15], *b2 = (const float*)d_in[16],
              *g2 = (const float*)d_in[17], *be2 = (const float*)d_in[18],
              *m2 = (const float*)d_in[19], *v2 = (const float*)d_in[20];
  char* ws = (char*)d_ws;
  float* out = (float*)d_out;

  prep_kernel<<<1, 256, 0, stream>>>(w0, b0, g0, be0, m0, v0, w1, b1, g1, be1, m1,
                                     v1, w2, b2, g2, be2, m2, v2, ws);
  fps_kernel<<<BATCH, 1024, 0, stream>>>(xyz, seed, out, (float*)(ws + WS_CENT));
  ball_kernel<<<(BATCH * NPOINT) / 8, 256, 0, stream>>>(
      xyz, pts, (const float*)(ws + WS_CENT), (_Float16*)(ws + WS_FEAT));
  mlp_kernel<<<(BATCH * NPOINT) / 4, 128, 0, stream>>>(
      (const _Float16*)(ws + WS_W0), (const float*)(ws + WS_BIAS),
      (const _Float16*)(ws + WS_FEAT), out);
}
